// EpsilonModel_16973710753852
// MI455X (gfx1250) — matrix operand order repaired and checked by reading, not compiled, not run on hardware
//
#include <hip/hip_runtime.h>
#include <hip/hip_bf16.h>
#include <float.h>

#define BB 32
#define SS 1024
#define DMODEL 128
#define DINNER 512
#define DSTATE 16
#define DCONV 8
#define DTRANK 8
#define NLAYERS 3
#define MROWS (BB * SS) /* 32768 */
#define GEMM_WAVES 4
#define NTILES 4 /* N-tiles (16 cols each) per wave */
#define NDBL (DTRANK + 2 * DSTATE) /* 40 */

typedef __attribute__((ext_vector_type(16))) __bf16 v16bf;
typedef __attribute__((ext_vector_type(8))) __bf16 v8bf;
typedef __attribute__((ext_vector_type(8))) float v8f;

// ---------------------------------------------------------------------------
// f32 -> bf16 elementwise conversion (weight pre-conversion)
// ---------------------------------------------------------------------------
__global__ void cvt_bf16_kernel(const float* __restrict__ in, __bf16* __restrict__ out,
                                size_t n) {
    size_t idx = (size_t)blockIdx.x * blockDim.x + threadIdx.x;
    if (idx < n) out[idx] = (__bf16)in[idx];
}

// ---------------------------------------------------------------------------
// start_max = max over x[:,:,2]
// ---------------------------------------------------------------------------
__global__ void reduce_max_stage1(const float* __restrict__ x, float* __restrict__ red) {
    __shared__ float sm[256];
    int tid = threadIdx.x;
    int idx = blockIdx.x * 256 + tid; // 128 blocks * 256 = 32768 elements
    float v = -FLT_MAX;
    if (idx < MROWS) v = x[(size_t)idx * 4 + 2];
    sm[tid] = v;
    __syncthreads();
    for (int off = 128; off > 0; off >>= 1) {
        if (tid < off) sm[tid] = fmaxf(sm[tid], sm[tid + off]);
        __syncthreads();
    }
    if (tid == 0) red[blockIdx.x] = sm[0];
}

__global__ void reduce_max_stage2(const float* __restrict__ red, float* __restrict__ smax) {
    __shared__ float sm[128];
    int tid = threadIdx.x; // 128 threads
    sm[tid] = red[tid];
    __syncthreads();
    for (int off = 64; off > 0; off >>= 1) {
        if (tid < off) sm[tid] = fmaxf(sm[tid], sm[tid + off]);
        __syncthreads();
    }
    if (tid == 0) smax[0] = sm[0];
}

// ---------------------------------------------------------------------------
// h_bf[b,s,d] = bf16( sum_k xn[b,s,k] * fc_w[d,k] + fc_b[d] )
// ---------------------------------------------------------------------------
__global__ void norm_fc_kernel(const float* __restrict__ x, const float* __restrict__ fc_w,
                               const float* __restrict__ fc_b, const float* __restrict__ smax,
                               __bf16* __restrict__ h_bf) {
    size_t idx = (size_t)blockIdx.x * blockDim.x + threadIdx.x; // MROWS*128
    if (idx >= (size_t)MROWS * DMODEL) return;
    int d = (int)(idx & (DMODEL - 1));
    size_t bs = idx >> 7;
    const float* xp = x + bs * 4;
    float inv_max = 1.0f / smax[0];
    float xn0 = xp[0] * (1.0f / 255.0f);
    float xn1 = xp[1] * (1.0f / 255.0f);
    float xn2 = xp[2] * inv_max;
    float xn3 = xp[3];
    const float* wr = fc_w + (size_t)d * 4;
    h_bf[idx] = (__bf16)(xn0 * wr[0] + xn1 * wr[1] + xn2 * wr[2] + xn3 * wr[3] + fc_b[d]);
}

// ---------------------------------------------------------------------------
// WMMA bf16 GEMM: C[M,N] = act(A[M,K] @ W[N,K]^T + bias)
// A and W are pre-converted bf16 (row-major). act: 0=none, 1=tanh, 2=relu.
// outmode bit0: write f32 Cf; bit1: write bf16 Cb.
// Each wave computes a 16x64 output strip: A fragment loaded once per k-step,
// 4 B fragments + 4 WMMAs. B rows clamped so the k-loop is branch-free;
// stores guarded by n < N.
// ---------------------------------------------------------------------------
__device__ __forceinline__ v8f wmma_bf16(v16bf a, v16bf b, v8f c) {
    return __builtin_amdgcn_wmma_f32_16x16x32_bf16(false, a, false, b, (short)0, c,
                                                   false, false);
}

__global__ void gemm_wmma(const __bf16* __restrict__ A, const __bf16* __restrict__ W,
                          const float* __restrict__ bias, float* __restrict__ Cf,
                          __bf16* __restrict__ Cb, int M, int N, int K, int act,
                          int outmode) {
    int lane = threadIdx.x & 31;
    int wave = threadIdx.x >> 5;
    int tileM = blockIdx.x * GEMM_WAVES + wave;
    int half = lane >> 4;
    int l15 = lane & 15;
    int m = tileM * 16 + l15;            // A row this lane loads
    int n0 = blockIdx.y * (16 * NTILES); // first column of this wave's strip

    // Per-tile B row pointers (clamped to a valid row; no branches in k-loop)
    const __bf16* wrow_base[NTILES];
    int ncol[NTILES];
#pragma unroll
    for (int t = 0; t < NTILES; ++t) {
        int n = n0 + t * 16 + l15;
        ncol[t] = n;
        int nld = (n < N) ? n : (N - 1);
        wrow_base[t] = W + (size_t)nld * K + half * 16;
    }

    v8f acc[NTILES];
#pragma unroll
    for (int t = 0; t < NTILES; ++t) acc[t] = (v8f){0.f, 0.f, 0.f, 0.f, 0.f, 0.f, 0.f, 0.f};

    const __bf16* arow_base = A + (size_t)m * K + half * 16;

    for (int kk = 0; kk < K; kk += 32) {
        const __bf16* arow = arow_base + kk;
        __builtin_prefetch(arow + 32, 0, 1); // next k-step of the A stream
        // A fragment: element i -> k = i + 16*half => 16 contiguous bf16
        v16bf a = *(const v16bf*)(arow);
#pragma unroll
        for (int t = 0; t < NTILES; ++t) {
            // B fragment: element i -> k = i + 16*half => 16 contiguous bf16
            v16bf b = *(const v16bf*)(wrow_base[t] + kk);
            acc[t] = wmma_bf16(a, b, acc[t]);
        }
    }

#pragma unroll
    for (int t = 0; t < NTILES; ++t) {
        int n = ncol[t];
        if (n >= N) continue;
        float bval = (bias != nullptr) ? bias[n] : 0.0f;
#pragma unroll
        for (int v = 0; v < 8; ++v) {
            int row = tileM * 16 + v + half * 8; // C layout: VGPR v -> M = v + 8*half
            float val = acc[t][v] + bval;
            if (act == 1) val = tanhf(val);
            else if (act == 2) val = fmaxf(val, 0.0f);
            if (outmode & 1) Cf[(size_t)row * N + n] = val;
            if (outmode & 2) Cb[(size_t)row * N + n] = (__bf16)val;
        }
    }
}

// ---------------------------------------------------------------------------
// Depthwise causal conv (8 taps) + bias + silu.
// Writes f32 (for the scan) and a bf16 shadow (for the x_proj WMMA GEMM).
// ---------------------------------------------------------------------------
__global__ void conv_silu_kernel(const float* __restrict__ xz, const float* __restrict__ cw,
                                 const float* __restrict__ cb, float* __restrict__ xi,
                                 __bf16* __restrict__ xi_bf) {
    size_t idx = (size_t)blockIdx.x * blockDim.x + threadIdx.x; // MROWS*512
    if (idx >= (size_t)MROWS * DINNER) return;
    int c = (int)(idx & (DINNER - 1));
    size_t bs = idx >> 9;
    int s = (int)(bs & (SS - 1));
    int b = (int)(bs >> 10);
    float acc = cb[c];
    const float* wr = cw + (size_t)c * DCONV;
#pragma unroll
    for (int j = 0; j < DCONV; ++j) {
        int sp = s - (DCONV - 1) + j;
        float xv = (sp >= 0) ? xz[(((size_t)b * SS + sp) * (2 * DINNER)) + c] : 0.0f;
        acc += xv * wr[j];
    }
    float sig = 1.0f / (1.0f + __expf(-acc));
    float r = acc * sig;
    xi[idx] = r;
    xi_bf[idx] = (__bf16)r;
}

// ---------------------------------------------------------------------------
// dt[b,s,d] = softplus( dbl[b,s,:8] @ dt_proj_w[d,:] + dt_proj_b[d] )
// ---------------------------------------------------------------------------
__global__ void dt_kernel(const float* __restrict__ dbl, const float* __restrict__ dtw,
                          const float* __restrict__ dtb, float* __restrict__ dt) {
    size_t idx = (size_t)blockIdx.x * blockDim.x + threadIdx.x; // MROWS*512
    if (idx >= (size_t)MROWS * DINNER) return;
    int d = (int)(idx & (DINNER - 1));
    size_t bs = idx >> 9;
    const float* row = dbl + bs * NDBL;
    const float* wr = dtw + (size_t)d * DTRANK;
    float acc = dtb[d];
#pragma unroll
    for (int r = 0; r < DTRANK; ++r) acc += row[r] * wr[r];
    float sp = (acc > 20.0f) ? acc : log1pf(__expf(acc));
    dt[idx] = sp;
}

// ---------------------------------------------------------------------------
// Selective scan: one block per batch, one thread per channel d.
// State (16 floats) lives in registers; B_t/C_t broadcast through LDS.
// Reads xi (f32), writes y as bf16 (only consumer is the out_proj WMMA GEMM).
// ---------------------------------------------------------------------------
__global__ void mamba_scan_kernel(const float* __restrict__ dt, const float* __restrict__ dbl,
                                  const float* __restrict__ xz, const float* __restrict__ A_log,
                                  const float* __restrict__ Dp, const float* __restrict__ xi,
                                  __bf16* __restrict__ y_bf) {
    int b = blockIdx.x;  // 0..31
    int d = threadIdx.x; // 0..511
    __shared__ float sB[DSTATE];
    __shared__ float sC[DSTATE];

    float Arow[DSTATE];
#pragma unroll
    for (int n = 0; n < DSTATE; ++n) Arow[n] = -__expf(A_log[(size_t)d * DSTATE + n]);
    float Dv = Dp[d];
    float st[DSTATE];
#pragma unroll
    for (int n = 0; n < DSTATE; ++n) st[n] = 0.0f;

    for (int s = 0; s < SS; ++s) {
        size_t base = (size_t)b * SS + s;
        if (d < 2 * DSTATE) {
            const float* p = dbl + base * NDBL + DTRANK;
            if (d < DSTATE) sB[d] = p[d];
            else sC[d - DSTATE] = p[d];
        }
        __syncthreads();
        float dtv = dt[base * DINNER + d];
        float xv = xi[base * DINNER + d];
        float zv = xz[base * (2 * DINNER) + DINNER + d];
        float scale = dtv * xv;
        float acc = 0.0f;
#pragma unroll
        for (int n = 0; n < DSTATE; ++n) {
            float dA = __expf(dtv * Arow[n]);
            st[n] = dA * st[n] + scale * sB[n];
            acc += st[n] * sC[n];
        }
        float sig = 1.0f / (1.0f + __expf(-zv));
        y_bf[base * DINNER + d] = (__bf16)((acc + Dv * xv) * (zv * sig));
        __syncthreads();
    }
}

// ---------------------------------------------------------------------------
// Head: t = leaky(h[:, -1, :] @ w^T + b), chained.  act slope = 0.01.
// ---------------------------------------------------------------------------
__global__ void head_fc_kernel(const float* __restrict__ in, const float* __restrict__ w,
                               const float* __restrict__ bias, float* __restrict__ out,
                               int in_dim, int out_dim, int in_stride, int in_offset) {
    int idx = blockIdx.x * blockDim.x + threadIdx.x; // BB*out_dim
    if (idx >= BB * out_dim) return;
    int o = idx % out_dim;
    int b = idx / out_dim;
    const float* row = in + (size_t)b * in_stride + in_offset;
    const float* wr = w + (size_t)o * in_dim;
    float acc = bias[o];
    for (int k = 0; k < in_dim; ++k) acc += row[k] * wr[k];
    out[idx] = (acc > 0.0f) ? acc : 0.01f * acc;
}

__global__ void head_final_kernel(const float* __restrict__ t2, const float* __restrict__ w3,
                                  const float* __restrict__ b3, const float* __restrict__ smax,
                                  float* __restrict__ out) {
    int idx = blockIdx.x * blockDim.x + threadIdx.x; // BB*2
    if (idx >= BB * 2) return;
    int j = idx % 2;
    int b = idx / 2;
    const float* row = t2 + (size_t)b * 512;
    const float* wr = w3 + (size_t)j * 512;
    float acc = b3[j];
    for (int k = 0; k < 512; ++k) acc += row[k] * wr[k];
    if (j == 0) acc *= smax[0];
    out[idx] = fmaxf(acc, 0.0f);
}

// ---------------------------------------------------------------------------
extern "C" void kernel_launch(void* const* d_in, const int* in_sizes, int n_in,
                              void* d_out, int out_size, void* d_ws, size_t ws_size,
                              hipStream_t stream) {
    const float* x = (const float*)d_in[0];
    const float* fc_w = (const float*)d_in[1];
    const float* fc_b = (const float*)d_in[2];
    const float* lin_w = (const float*)d_in[3];
    const float* lin_b = (const float*)d_in[4];
    const float* in_proj_w = (const float*)d_in[5];
    const float* conv_w = (const float*)d_in[6];
    const float* conv_b = (const float*)d_in[7];
    const float* x_proj_w = (const float*)d_in[8];
    const float* dt_proj_w = (const float*)d_in[9];
    const float* dt_proj_b = (const float*)d_in[10];
    const float* A_log = (const float*)d_in[11];
    const float* Dp = (const float*)d_in[12];
    const float* out_proj_w = (const float*)d_in[13];
    const float* w1 = (const float*)d_in[14];
    const float* b1 = (const float*)d_in[15];
    const float* w2 = (const float*)d_in[16];
    const float* b2 = (const float*)d_in[17];
    const float* w3 = (const float*)d_in[18];
    const float* b3 = (const float*)d_in[19];
    float* out = (float*)d_out;

    // ---- Workspace carve-up (units: floats; bf16 regions use half-floats) --
    float* w = (float*)d_ws;
    float* red = w;                                          // 256
    float* smax = w + 256;                                   // 1 (pad to 320)
    float* hA = w + 320;                                     // MROWS*128 f32
    float* xz = hA + (size_t)MROWS * DMODEL;                 // MROWS*1024 f32
    float* xi = xz + (size_t)MROWS * 2 * DINNER;             // MROWS*512 f32
    float* dbl = xi + (size_t)MROWS * DINNER;                // MROWS*40 f32
    float* dt = dbl + (size_t)MROWS * NDBL;                  // MROWS*512 f32
    float* t1 = dt + (size_t)MROWS * DINNER;                 // BB*512
    float* t2 = t1 + (size_t)BB * 512;                       // BB*512
    __bf16* hA_bf = (__bf16*)(t2 + (size_t)BB * 512);        // MROWS*128 bf16
    __bf16* hB_bf = hA_bf + (size_t)MROWS * DMODEL;          // MROWS*128 bf16
    __bf16* xi_bf = hB_bf + (size_t)MROWS * DMODEL;          // MROWS*512 bf16 (xi, then y)
    __bf16* wlin_bf = xi_bf + (size_t)MROWS * DINNER;        // 3*128*128
    __bf16* wip_bf = wlin_bf + (size_t)NLAYERS * DMODEL * DMODEL;      // 3*1024*128
    __bf16* wxp_bf = wip_bf + (size_t)NLAYERS * 2 * DINNER * DMODEL;   // 3*40*512
    __bf16* wop_bf = wxp_bf + (size_t)NLAYERS * NDBL * DINNER;         // 3*128*512

    // ---- Pre-convert all GEMM weights to bf16 (once per call) --------------
    {
        size_t n;
        n = (size_t)NLAYERS * DMODEL * DMODEL;
        cvt_bf16_kernel<<<(unsigned)((n + 255) / 256), 256, 0, stream>>>(lin_w, wlin_bf, n);
        n = (size_t)NLAYERS * 2 * DINNER * DMODEL;
        cvt_bf16_kernel<<<(unsigned)((n + 255) / 256), 256, 0, stream>>>(in_proj_w, wip_bf, n);
        n = (size_t)NLAYERS * NDBL * DINNER;
        cvt_bf16_kernel<<<(unsigned)((n + 255) / 256), 256, 0, stream>>>(x_proj_w, wxp_bf, n);
        n = (size_t)NLAYERS * DMODEL * DINNER;
        cvt_bf16_kernel<<<(unsigned)((n + 255) / 256), 256, 0, stream>>>(out_proj_w, wop_bf, n);
    }

    // ---- start_max reduction ----------------------------------------------
    reduce_max_stage1<<<128, 256, 0, stream>>>(x, red);
    reduce_max_stage2<<<1, 128, 0, stream>>>(red, smax);

    // ---- normalize + fc -> hA_bf ------------------------------------------
    {
        size_t n = (size_t)MROWS * DMODEL;
        norm_fc_kernel<<<(unsigned)((n + 255) / 256), 256, 0, stream>>>(x, fc_w, fc_b, smax,
                                                                        hA_bf);
    }

    dim3 blk(32 * GEMM_WAVES);
    unsigned gx = MROWS / (16 * GEMM_WAVES); // 512
    const int STRIP = 16 * NTILES;           // 64 columns per wave

    for (int i = 0; i < NLAYERS; ++i) {
        const float* lb = lin_b + (size_t)i * DMODEL;
        const float* cw = conv_w + (size_t)i * DINNER * DCONV;
        const float* cb = conv_b + (size_t)i * DINNER;
        const float* dtw = dt_proj_w + (size_t)i * DINNER * DTRANK;
        const float* dtb = dt_proj_b + (size_t)i * DINNER;
        const float* al = A_log + (size_t)i * DINNER * DSTATE;
        const float* dp = Dp + (size_t)i * DINNER;
        const __bf16* lwb = wlin_bf + (size_t)i * DMODEL * DMODEL;
        const __bf16* ipwb = wip_bf + (size_t)i * 2 * DINNER * DMODEL;
        const __bf16* xpwb = wxp_bf + (size_t)i * NDBL * DINNER;
        const __bf16* opwb = wop_bf + (size_t)i * DMODEL * DINNER;

        // h_mid = tanh(h @ lin_w^T + lin_b) -> hB_bf   (N=128, K=128)
        gemm_wmma<<<dim3(gx, DMODEL / STRIP), blk, 0, stream>>>(
            hA_bf, lwb, lb, nullptr, hB_bf, MROWS, DMODEL, DMODEL, 1, 2);
        // xz = h_mid @ in_proj_w^T (f32)               (N=1024, K=128)
        gemm_wmma<<<dim3(gx, (2 * DINNER) / STRIP), blk, 0, stream>>>(
            hB_bf, ipwb, nullptr, xz, nullptr, MROWS, 2 * DINNER, DMODEL, 0, 1);
        // xi = silu(conv(xz[:, :512]) + conv_b)  (f32 + bf16 shadow)
        {
            size_t n = (size_t)MROWS * DINNER;
            conv_silu_kernel<<<(unsigned)((n + 255) / 256), 256, 0, stream>>>(xz, cw, cb, xi,
                                                                              xi_bf);
        }
        // dbl = xi @ x_proj_w^T (f32)                  (N=40, K=512)
        gemm_wmma<<<dim3(gx, 1), blk, 0, stream>>>(
            xi_bf, xpwb, nullptr, dbl, nullptr, MROWS, NDBL, DINNER, 0, 1);
        // dt = softplus(dbl[:, :8] @ dt_proj_w^T + dt_proj_b)
        {
            size_t n = (size_t)MROWS * DINNER;
            dt_kernel<<<(unsigned)((n + 255) / 256), 256, 0, stream>>>(dbl, dtw, dtb, dt);
        }
        // selective scan; y (bf16) overwrites xi_bf
        mamba_scan_kernel<<<BB, DINNER, 0, stream>>>(dt, dbl, xz, al, dp, xi, xi_bf);
        // h = relu(y @ out_proj_w^T) -> hA (f32) + hA_bf (bf16)  (N=128, K=512)
        gemm_wmma<<<dim3(gx, DMODEL / STRIP), blk, 0, stream>>>(
            xi_bf, opwb, nullptr, hA, hA_bf, MROWS, DMODEL, DINNER, 2, 3);
    }

    // ---- Head MLP on h[:, -1, :] ------------------------------------------
    head_fc_kernel<<<(BB * 512 + 255) / 256, 256, 0, stream>>>(
        hA, w1, b1, t1, DMODEL, 512, SS * DMODEL, (SS - 1) * DMODEL);
    head_fc_kernel<<<(BB * 512 + 255) / 256, 256, 0, stream>>>(
        t1, w2, b2, t2, 512, 512, 512, 0);
    head_final_kernel<<<1, 64, 0, stream>>>(t2, w3, b3, smax, out);
}
